// XformersFavorXGLMAttention_32847909880068
// MI455X (gfx1250) — matrix operand order repaired and checked by reading, not compiled, not run on hardware
//
#include <hip/hip_runtime.h>
#include <stdint.h>

typedef __attribute__((ext_vector_type(16))) _Float16 v16h;
typedef __attribute__((ext_vector_type(8)))  _Float16 v8h;
typedef __attribute__((ext_vector_type(4)))  _Float16 v4h;
typedef __attribute__((ext_vector_type(8)))  float    v8f;

static constexpr int Dm  = 1024;   // model dim
static constexpr int Ss  = 1024;   // sequence length
static constexpr int Bb  = 8;      // batch
static constexpr int Hh  = 16;     // heads
static constexpr int HD  = 64;     // head dim
static constexpr int KL  = 256;    // Linformer projected length
static constexpr int BS  = Bb * Ss;   // 8192 rows
static constexpr int BH  = Bb * Hh;   // 128 (b,h) pairs

// ---- WMMA primitive: D = A(16x32 f16) x B(32x16 f16) + C(16x16 f32) ----
__device__ __forceinline__ v8f wmma16(v16h a, v16h b, v8f c) {
  return __builtin_amdgcn_wmma_f32_16x16x32_f16(
      /*neg_a=*/false, a, /*neg_b=*/false, b,
      /*c_mod=*/(short)0, c, /*reuse_a=*/false, /*reuse_b=*/false);
}

// A-operand fragment (16-bit A 16x32), contiguous k order to match B:
//   lanes 0-15 hold K = [k0..k0+15]
//   lanes16-31 hold K = [k0+16..k0+31]
// caller passes p = row_ptr + k0 + (lane>>4)*8; chunks at p+(lane>>4)*8 and +8.
__device__ __forceinline__ v16h load_a16(const _Float16* p) {
  p += ((threadIdx.x & 31) >> 4) * 8;
  v8h lo = *(const v8h*)(p);
  v8h hi = *(const v8h*)(p + 8);
  v16h r;
#pragma unroll
  for (int i = 0; i < 8; ++i) { r[i] = lo[i]; r[i + 8] = hi[i]; }
  return r;
}
// B-operand (32x16): lane n = column (n&15); lanes 0-15 hold K=0..15,
// lanes 16-31 hold K=16..31 -> one contiguous 32B v16h load per lane when the
// B column is contiguous in memory (k-major).

// Async global->LDS copy of 16 bytes per lane (CDNA5, ASYNCcnt-tracked).
__device__ __forceinline__ void async_copy_b128(uint32_t lds_off, const void* gsrc) {
  asm volatile("global_load_async_to_lds_b128 %0, %1, off"
               :: "v"(lds_off), "v"((uint64_t)(uintptr_t)gsrc)
               : "memory");
}
__device__ __forceinline__ void wait_async_le1() {
  asm volatile("s_wait_asynccnt 0x1" ::: "memory");
}
__device__ __forceinline__ void wait_async_0() {
  asm volatile("s_wait_asynccnt 0x0" ::: "memory");
}

// ---------------- fp32 -> fp16 conversion ----------------
__global__ void cvt_f2h(const float* __restrict__ src,
                        _Float16* __restrict__ dst, int n) {
  int i = (blockIdx.x * blockDim.x + threadIdx.x) * 4;
  int stride = gridDim.x * blockDim.x * 4;
  for (; i < n; i += stride) {
    float4 v = *(const float4*)(src + i);
    v4h h;
    h[0] = (_Float16)v.x; h[1] = (_Float16)v.y;
    h[2] = (_Float16)v.z; h[3] = (_Float16)v.w;
    *(v4h*)(dst + i) = h;
  }
}

// ---------------- C[M,N] = A[M,K] * W[N,K]^T + bias ----------------
// Block = 8 waves sharing one 64-column N strip. The 64x32 W tile for each
// k-step is async-copied (global_load_async_to_lds_b128, double-buffered)
// into LDS once per block; every wave reads B fragments from LDS.
template <bool OUT_F32>
__global__ __launch_bounds__(256) void gemm_xwT(
    const _Float16* __restrict__ A, const _Float16* __restrict__ W,
    const float* __restrict__ bias, void* __restrict__ outp,
    int M, int N, int K) {
  const int wid = threadIdx.x >> 5;
  const int lane = threadIdx.x & 31;
  const int hi = lane >> 4, ln = lane & 15;
  const int m_blocks = M >> 7;                 // 8 m-tiles per block
  const int ni = blockIdx.x / m_blocks;        // N strip (64 cols)
  const int mi = (blockIdx.x % m_blocks) * 8 + wid;

  __shared__ __align__(32) _Float16 wtile[2][64][32];  // [buf][col][k] 8 KB

  // per-thread slot of the cooperative W-tile copy: 256 thr x 16 B = 4 KB
  const int col_c = threadIdx.x >> 2;          // 0..63
  const int kc_c  = (threadIdx.x & 3) * 8;     // 0,8,16,24
  const _Float16* wsrc = W + (size_t)(ni * 64 + col_c) * K + kc_c;

  const _Float16* arow = A + (size_t)(mi * 16 + ln) * K;
  v8f acc[4] = {};

  const int iters = K >> 5;
  // prologue: tile 0 in flight
  async_copy_b128((uint32_t)(uintptr_t)&wtile[0][col_c][kc_c], wsrc);

  for (int it = 0; it < iters; ++it) {
    const int buf = it & 1;
    if (it + 1 < iters) {
      async_copy_b128((uint32_t)(uintptr_t)&wtile[buf ^ 1][col_c][kc_c],
                      wsrc + (size_t)(it + 1) * 32);
      wait_async_le1();   // tile `it` landed; tile `it+1` still in flight
    } else {
      wait_async_0();
    }
    __syncthreads();

    const int k0 = it * 32;
    v16h a = load_a16(arow + k0 + hi * 8);
    __builtin_prefetch(arow + k0 + 256, 0, 0);  // global_prefetch_b8
#pragma unroll
    for (int nt = 0; nt < 4; ++nt) {
      v16h b = *(const v16h*)&wtile[buf][nt * 16 + ln][hi * 16];
      acc[nt] = wmma16(a, b, acc[nt]);
    }
    __syncthreads();
  }

#pragma unroll
  for (int nt = 0; nt < 4; ++nt) {
    int col = ni * 64 + nt * 16 + ln;
    float bv = bias[col];
#pragma unroll
    for (int r = 0; r < 8; ++r) {
      size_t idx = (size_t)(mi * 16 + r + hi * 8) * N + col;
      float v = acc[nt][r] + bv;
      if (OUT_F32) ((float*)outp)[idx] = v;
      else         ((_Float16*)outp)[idx] = (_Float16)v;
    }
  }
}

// ------- kp/vp: out[bh][kidx][d] = sum_s Pw[kidx,s] * Sin[b,s,h*64+d] -------
// One block per (b,h). LDS stages a transposed 32(s) x 64(d) tile of K/V so
// B-fragments are contiguous LDS vectors. vp stored transposed [d][k] so the
// attention AV GEMM gets contiguous B loads.
__global__ __launch_bounds__(256) void proj_seq(
    const _Float16* __restrict__ Sin, const _Float16* __restrict__ Pw,
    _Float16* __restrict__ outb, int transpose_out) {
  const int bh = blockIdx.x;
  const int b = bh >> 4, h = bh & 15;
  const int wid = threadIdx.x >> 5, lane = threadIdx.x & 31;
  const int hi = lane >> 4, ln = lane & 15;
  __shared__ __align__(64) _Float16 tileT[64][32];  // [d][s]

  v8f acc[2][4] = {};
  const _Float16* sbase = Sin + (size_t)b * Ss * Dm + h * HD;
  const int sl = threadIdx.x >> 3;       // 0..31  (s within tile)
  const int dc = (threadIdx.x & 7) * 8;  // 0..56  (d chunk)

  for (int s0 = 0; s0 < Ss; s0 += 32) {
    v8h v = *(const v8h*)(sbase + (size_t)(s0 + sl) * Dm + dc);
#pragma unroll
    for (int j = 0; j < 8; ++j) tileT[dc + j][sl] = v[j];
    __syncthreads();
#pragma unroll
    for (int mt = 0; mt < 2; ++mt) {
      int kidx = wid * 32 + mt * 16 + ln;
      v16h a = load_a16(Pw + (size_t)kidx * Ss + s0 + hi * 8);
#pragma unroll
      for (int nt = 0; nt < 4; ++nt) {
        v16h bf = *(const v16h*)&tileT[nt * 16 + ln][hi * 16];  // ds_load_b128 x2
        acc[mt][nt] = wmma16(a, bf, acc[mt][nt]);
      }
    }
    __syncthreads();
  }
#pragma unroll
  for (int mt = 0; mt < 2; ++mt)
#pragma unroll
    for (int nt = 0; nt < 4; ++nt)
#pragma unroll
      for (int r = 0; r < 8; ++r) {
        int kidx = wid * 32 + mt * 16 + r + hi * 8;
        int d = nt * 16 + ln;
        _Float16 v = (_Float16)acc[mt][nt][r];
        if (!transpose_out)
          outb[(size_t)bh * KL * HD + (size_t)kidx * HD + d] = v;
        else
          outb[(size_t)bh * KL * HD + (size_t)d * KL + kidx] = v;
      }
}

// ------- fused attention: scores -> softmax -> attn @ vp, per 16-row tile ----
__global__ __launch_bounds__(128) void linattn(
    const _Float16* __restrict__ Q, const _Float16* __restrict__ kp,
    const _Float16* __restrict__ vpT, _Float16* __restrict__ Y) {
  __shared__ float sc[4][16 * KL];  // 64 KB, one 16x256 score tile per wave
  const int wid = threadIdx.x >> 5, lane = threadIdx.x & 31;
  const int hi = lane >> 4, ln = lane & 15;
  const int t = blockIdx.x * 4 + wid;         // global s-tile index
  const int bh = t >> 6;                      // 64 tiles per (b,h)
  const int s0 = (t & 63) * 16;
  const int b = bh >> 4, h = bh & 15;
  float* ws = sc[wid];

  const _Float16* qbase = Q + (size_t)(b * Ss + s0) * Dm + h * HD;
  const _Float16* kpb = kp + (size_t)bh * KL * HD;

  // Phase 1: scores = (Q @ kp^T) / sqrt(64)
  for (int nt = 0; nt < 16; ++nt) {
    v8f acc = {};
#pragma unroll
    for (int k0 = 0; k0 < HD; k0 += 32) {
      v16h a = load_a16(qbase + (size_t)ln * Dm + k0 + hi * 8);
      v16h bf = *(const v16h*)(kpb + (size_t)(nt * 16 + ln) * HD + k0 + hi * 16);
      acc = wmma16(a, bf, acc);
    }
#pragma unroll
    for (int r = 0; r < 8; ++r)
      ws[(r + hi * 8) * KL + nt * 16 + ln] = acc[r] * 0.125f;
  }
  __syncthreads();

  // Phase 2: softmax over 256, two lanes per row (lane pair L, L^16)
  float* rp = ws + ln * KL + hi * 128;
  float m = -3.0e38f;
  for (int i = 0; i < 128; ++i) m = fmaxf(m, rp[i]);
  m = fmaxf(m, __shfl_xor(m, 16));
  float sum = 0.f;
  for (int i = 0; i < 128; ++i) { float e = __expf(rp[i] - m); rp[i] = e; sum += e; }
  sum += __shfl_xor(sum, 16);
  float rinv = 1.0f / sum;  // lane L holds 1/sum for row L&15 (= its A-frag row)
  __syncthreads();

  // Phase 3: y = attn @ vp   (vp stored [d][k] -> contiguous B loads)
  const _Float16* vb = vpT + (size_t)bh * KL * HD;
  v8f yacc[4] = {};
  for (int k0 = 0; k0 < KL; k0 += 32) {
    const float* ar = ws + ln * KL + k0 + hi * 16;
    v16h a;
#pragma unroll
    for (int i = 0; i < 8; ++i) {
      a[i]     = (_Float16)(ar[i]      * rinv);
      a[i + 8] = (_Float16)(ar[i + 8] * rinv);
    }
#pragma unroll
    for (int nt = 0; nt < 4; ++nt) {
      v16h bf = *(const v16h*)(vb + (size_t)(nt * 16 + ln) * KL + k0 + hi * 16);
      yacc[nt] = wmma16(a, bf, yacc[nt]);
    }
  }
  _Float16* yb = Y + (size_t)(b * Ss + s0) * Dm + h * HD;
#pragma unroll
  for (int nt = 0; nt < 4; ++nt)
#pragma unroll
    for (int r = 0; r < 8; ++r)
      yb[(size_t)(r + hi * 8) * Dm + nt * 16 + ln] = (_Float16)yacc[nt][r];
}

extern "C" void kernel_launch(void* const* d_in, const int* in_sizes, int n_in,
                              void* d_out, int out_size, void* d_ws, size_t ws_size,
                              hipStream_t stream) {
  (void)in_sizes; (void)n_in; (void)out_size; (void)ws_size;
  const float* X   = (const float*)d_in[0];
  const float* q_w = (const float*)d_in[1];
  const float* q_b = (const float*)d_in[2];
  const float* k_w = (const float*)d_in[3];
  const float* k_b = (const float*)d_in[4];
  const float* v_w = (const float*)d_in[5];
  const float* v_b = (const float*)d_in[6];
  const float* E_w = (const float*)d_in[7];
  const float* F_w = (const float*)d_in[8];
  const float* o_w = (const float*)d_in[9];
  const float* o_b = (const float*)d_in[10];

  char* ws = (char*)d_ws;
  size_t off = 0;
  auto take = [&](size_t elems) {
    _Float16* p = (_Float16*)(ws + off);
    off += elems * sizeof(_Float16);
    return p;
  };
  _Float16* Xh  = take((size_t)BS * Dm);
  _Float16* Qh  = take((size_t)BS * Dm);
  _Float16* Kh  = take((size_t)BS * Dm);
  _Float16* Vh  = take((size_t)BS * Dm);
  _Float16* Yh  = take((size_t)BS * Dm);
  _Float16* qwh = take((size_t)Dm * Dm);
  _Float16* kwh = take((size_t)Dm * Dm);
  _Float16* vwh = take((size_t)Dm * Dm);
  _Float16* owh = take((size_t)Dm * Dm);
  _Float16* Ewh = take((size_t)KL * Dm);
  _Float16* Fwh = take((size_t)KL * Dm);
  _Float16* kpb = take((size_t)BH * KL * HD);
  _Float16* vpb = take((size_t)BH * KL * HD);

  auto cvt = [&](const float* s, _Float16* d, int n) {
    int blocks = (n / 4 + 255) / 256;
    cvt_f2h<<<dim3(blocks), dim3(256), 0, stream>>>(s, d, n);
  };
  cvt(X,   Xh,  BS * Dm);
  cvt(q_w, qwh, Dm * Dm);
  cvt(k_w, kwh, Dm * Dm);
  cvt(v_w, vwh, Dm * Dm);
  cvt(o_w, owh, Dm * Dm);
  cvt(E_w, Ewh, KL * Dm);
  cvt(F_w, Fwh, KL * Dm);

  // (8192/128) m-blocks * (1024/64) n-strips = 1024 blocks of 8 waves
  dim3 gblk(1024);
  gemm_xwT<false><<<gblk, 256, 0, stream>>>(Xh, qwh, q_b, Qh, BS, Dm, Dm);
  gemm_xwT<false><<<gblk, 256, 0, stream>>>(Xh, kwh, k_b, Kh, BS, Dm, Dm);
  gemm_xwT<false><<<gblk, 256, 0, stream>>>(Xh, vwh, v_b, Vh, BS, Dm, Dm);

  proj_seq<<<dim3(BH), 256, 0, stream>>>(Kh, Ewh, kpb, 0);
  proj_seq<<<dim3(BH), 256, 0, stream>>>(Vh, Fwh, vpb, 1);

  // 128 (b,h) * 64 s-tiles = 8192 wave-tiles, 4 waves/block
  linattn<<<dim3(BH * 64 / 4), 128, 0, stream>>>(Qh, kpb, vpb, Yh);

  gemm_xwT<true><<<gblk, 256, 0, stream>>>(Yh, owh, o_b, d_out, BS, Dm, Dm);
}
